// myGRUCell_48266842472681
// MI455X (gfx1250) — matrix operand order repaired and checked by reading, not compiled, not run on hardware
//
#include <hip/hip_runtime.h>
#include <hip/hip_bf16.h>

// Fused GRU cell for MI455X (gfx1250, wave32).
// 5 GEMMs [16384x1024]x[1024x1024]^T fused with gate math.
// fp32 -> bf16 on the fly, v_wmma_f32_16x16x32_bf16 with fp32 accumulate.
// LDS double-buffered pipeline: stage tile k+1 while WMMAs consume tile k.

typedef __attribute__((ext_vector_type(16))) __bf16       v16bf;
typedef __attribute__((ext_vector_type(2)))  __bf16       v2bf;
typedef __attribute__((ext_vector_type(8)))  float        v8f;
typedef __attribute__((ext_vector_type(4)))  float        f4;
typedef __attribute__((ext_vector_type(4)))  unsigned int u4;
typedef __attribute__((ext_vector_type(2)))  unsigned int u2;

#define KDIM 1024            // I == H == 1024 (shared reduction dim)
#define HOUT 1024            // output width
#define BM 128               // batch rows per workgroup
#define BN 64                // output cols per workgroup
#define BK 32                // k-step == WMMA bf16 K
#define NTHREADS 256         // 8 wave32s: 4 (M) x 2 (N)
#define LDR 40               // LDS row stride in bf16 elems (32 + 8 pad -> conflict-free b128)
#define A_TILE (BM * LDR)    // 5120 elems
#define W_TILE (BN * LDR)    // 2560 elems
#define BUF_ELEMS (2 * A_TILE + 5 * W_TILE)   // 23040 elems = 46080 B per buffer

__device__ __forceinline__ unsigned short f2bf(float f) {
    // round-to-nearest-even f32 -> bf16 (fallback path)
    unsigned int u = __builtin_bit_cast(unsigned int, f);
    u += 0x7FFFu + ((u >> 16) & 1u);
    return (unsigned short)(u >> 16);
}

__device__ __forceinline__ unsigned int pk_bf16(float a, float b) {
#if __has_builtin(__builtin_amdgcn_cvt_pk_bf16_f32)
    v2bf r = __builtin_amdgcn_cvt_pk_bf16_f32(a, b);
    return __builtin_bit_cast(unsigned int, r);
#else
    return (unsigned int)f2bf(a) | ((unsigned int)f2bf(b) << 16);
#endif
}

union Frag {
    v16bf v;
    u4    q[2];
};

__device__ __forceinline__ float sigmoid_fast(float x) {
    return 1.0f / (1.0f + __expf(-x));
}

__device__ __forceinline__ float tanh_fast(float x) {
    float e = __expf(-2.0f * __builtin_fabsf(x));   // e in (0,1] -> no inf/inf
    float r = (1.0f - e) / (1.0f + e);
    return __builtin_copysignf(r, x);
}

// Stage one BK-slice of all 7 tiles: global f32 -> bf16 -> LDS.
// A tiles are BMx32 (1024 float4 each, 4/thread); W tiles BNx32 (512 float4, 2/thread).
__device__ __forceinline__ void stage_tiles(unsigned short* __restrict__ sb,
                                            const float* const* gsrc,
                                            int tid, int k0) {
#pragma unroll
    for (int t = 0; t < 2; ++t) {
        unsigned short* s = sb + t * A_TILE;
        const float* g = gsrc[t];
#pragma unroll
        for (int i = 0; i < 4; ++i) {
            int f = tid + NTHREADS * i;   // float4 index in BMx32 tile (8 f4/row)
            int row = f >> 3, c4 = f & 7;
            f4 v = *(const f4*)(g + (long long)row * KDIM + k0 + c4 * 4);
            u2 p;
            p.x = pk_bf16(v.x, v.y);
            p.y = pk_bf16(v.z, v.w);
            *(u2*)(s + row * LDR + c4 * 4) = p;
        }
    }
#pragma unroll
    for (int t = 0; t < 5; ++t) {
        unsigned short* s = sb + 2 * A_TILE + t * W_TILE;
        const float* g = gsrc[2 + t];
#pragma unroll
        for (int i = 0; i < 2; ++i) {
            int f = tid + NTHREADS * i;   // float4 index in BNx32 tile
            int row = f >> 3, c4 = f & 7;
            f4 v = *(const f4*)(g + (long long)row * KDIM + k0 + c4 * 4);
            u2 p;
            p.x = pk_bf16(v.x, v.y);
            p.y = pk_bf16(v.z, v.w);
            *(u2*)(s + row * LDR + c4 * 4) = p;
        }
    }
}

__global__ __launch_bounds__(NTHREADS) void gru_fused_wmma(
    const float* __restrict__ input,  const float* __restrict__ hidden,
    const float* __restrict__ Wri,    const float* __restrict__ Wzi,
    const float* __restrict__ Wni,    const float* __restrict__ Wrh,
    const float* __restrict__ Wnh,
    const float* __restrict__ bri,    const float* __restrict__ bzi,
    const float* __restrict__ bni,    const float* __restrict__ brh,
    const float* __restrict__ bnh,
    float* __restrict__ out, long long halfOff)
{
    // double-buffered: [0]=input(BMx32), [1]=hidden(BMx32), [2..6]=W tiles (BNx32)
    __shared__ unsigned short smem[2 * BUF_ELEMS];   // 90 KB of the 320 KB WGP pool

    const int tid  = threadIdx.x;
    const int lane = tid & 31;
    const int wave = tid >> 5;     // 8 waves: 4x2 grid of 32x32 sub-tiles
    const int wm   = wave >> 1;    // 0..3 (M)
    const int wn   = wave & 1;     // 0..1 (N)
    const int lr   = lane & 15;
    const int lh   = lane >> 4;

    const int rowBase = blockIdx.y * BM;   // batch rows
    const int colBase = blockIdx.x * BN;   // output cols == weight rows

    const float* gsrc[7] = {
        input  + (long long)rowBase * KDIM,
        hidden + (long long)rowBase * KDIM,
        Wri    + (long long)colBase * KDIM,
        Wzi    + (long long)colBase * KDIM,
        Wni    + (long long)colBase * KDIM,
        Wrh    + (long long)colBase * KDIM,
        Wnh    + (long long)colBase * KDIM };

    // acc[0]=x@Wri  acc[1]=x@Wzi  acc[2]=x@Wni  acc[3]=h@Wrh  acc[4]=h@Wnh
    v8f acc[5][2][2];
#pragma unroll
    for (int g = 0; g < 5; ++g)
#pragma unroll
        for (int i = 0; i < 2; ++i)
#pragma unroll
            for (int j = 0; j < 2; ++j) {
                v8f z = {};
                acc[g][i][j] = z;
            }

    int p = 0;
    stage_tiles(smem, gsrc, tid, 0);
    __syncthreads();

#pragma unroll 1
    for (int k0 = 0; k0 < KDIM; k0 += BK) {
        // prefetch next slice into the pong buffer while ping feeds the WMMAs
        if (k0 + BK < KDIM)
            stage_tiles(smem + (p ^ 1) * BUF_ELEMS, gsrc, tid, k0 + BK);

        const unsigned short* sb = smem + p * BUF_ELEMS;

        // A fragments (16x32 bf16, ISA lane layout): 2x ds_load_b128 each
        Frag aIn[2], aHid[2];
#pragma unroll
        for (int ms = 0; ms < 2; ++ms) {
            int r = wm * 32 + ms * 16 + lr;
            const unsigned short* pI = sb + r * LDR;
            const unsigned short* pH = sb + A_TILE + r * LDR;
            aIn[ms].q[0]  = *(const u4*)(pI + lh * 8);        // K 0..15 half
            aIn[ms].q[1]  = *(const u4*)(pI + 16 + lh * 8);   // K 16..31 half
            aHid[ms].q[0] = *(const u4*)(pH + lh * 8);
            aHid[ms].q[1] = *(const u4*)(pH + 16 + lh * 8);
        }

        // per gate: B fragments (32x16 bf16) + 2x2 WMMAs
#pragma unroll
        for (int g = 0; g < 5; ++g) {
            const unsigned short* wb = sb + 2 * A_TILE + g * W_TILE;
            Frag bfr[2];
#pragma unroll
            for (int ns = 0; ns < 2; ++ns) {
                int n = wn * 32 + ns * 16 + lr;
                const unsigned short* q = wb + n * LDR + lh * 8;
                bfr[ns].q[0] = *(const u4*)(q);
                bfr[ns].q[1] = *(const u4*)(q + 16);
            }
            const Frag* A = (g < 3) ? aIn : aHid;
#pragma unroll
            for (int ms = 0; ms < 2; ++ms)
#pragma unroll
                for (int ns = 0; ns < 2; ++ns)
                    acc[g][ms][ns] = __builtin_amdgcn_wmma_f32_16x16x32_bf16(
                        false, A[ms].v, false, bfr[ns].v,
                        (short)0, acc[g][ms][ns], false, false);
        }
        __syncthreads();
        p ^= 1;
    }

    // epilogue: gate math in registers, single write of next_hidden (twice: tuple)
#pragma unroll
    for (int ms = 0; ms < 2; ++ms) {
#pragma unroll
        for (int ns = 0; ns < 2; ++ns) {
            int n = colBase + wn * 32 + ns * 16 + lr;
            float v_bri = bri[n], v_bzi = bzi[n], v_bni = bni[n];
            float v_brh = brh[n], v_bnh = bnh[n];
#pragma unroll
            for (int d = 0; d < 8; ++d) {
                int m = rowBase + wm * 32 + ms * 16 + lh * 8 + d;
                float rh = acc[3][ms][ns][d] + v_brh;              // reused by r and z
                float r  = sigmoid_fast(acc[0][ms][ns][d] + v_bri + rh);
                float z  = sigmoid_fast(acc[1][ms][ns][d] + v_bzi + rh);
                float nn = tanh_fast(acc[2][ms][ns][d] + v_bni +
                                     r * (acc[4][ms][ns][d] + v_bnh));
                float hp = hidden[(long long)m * KDIM + n];
                float o  = (1.0f - z) * nn + z * hp;
                long long idx = (long long)m * HOUT + n;
                out[idx] = o;
                out[idx + halfOff] = o;
            }
        }
    }
}

extern "C" void kernel_launch(void* const* d_in, const int* in_sizes, int n_in,
                              void* d_out, int out_size, void* d_ws, size_t ws_size,
                              hipStream_t stream) {
    (void)n_in; (void)d_ws; (void)ws_size;
    const float* input  = (const float*)d_in[0];
    const float* hidden = (const float*)d_in[1];
    const float* Wri = (const float*)d_in[2];   const float* bri = (const float*)d_in[3];
    const float* Wrh = (const float*)d_in[4];   const float* brh = (const float*)d_in[5];
    const float* Wzi = (const float*)d_in[6];   const float* bzi = (const float*)d_in[7];
    // d_in[8] = W_z_h, d_in[9] = b_z_h: dead in the reference (z reuses r_h)
    const float* Wni = (const float*)d_in[10];  const float* bni = (const float*)d_in[11];
    const float* Wnh = (const float*)d_in[12];  const float* bnh = (const float*)d_in[13];
    float* out = (float*)d_out;

    int batch = in_sizes[0] / KDIM;                 // 16384
    long long halfOff = (long long)out_size / 2;    // tuple (h, h) concatenated

    dim3 grid(HOUT / BN, batch / BM);               // (16, 128)
    gru_fused_wmma<<<grid, NTHREADS, 0, stream>>>(input, hidden,
                                                  Wri, Wzi, Wni, Wrh, Wnh,
                                                  bri, bzi, bni, brh, bnh,
                                                  out, halfOff);
}
